// CliffordFieldBlock_65910568125113
// MI455X (gfx1250) — matrix operand order repaired and checked by reading, not compiled, not run on hardware
//
#include <hip/hip_runtime.h>
#include <math.h>

// ---------------------------------------------------------------------------
// CliffordFieldBlock forward for MI455X (gfx1250, wave32, WMMA f16->f32)
// B=8, H=W=32, D=64, S=1024, R=B*S=8192, G=4
// A-fragments: 2x ds_load_b128 from row-major LDS tiles.
// B-fragments: 2x ds_load_b128 from swizzled-B LDS tiles.
// Transcendentals: native v_tanh_f32 / v_exp_f32 paths.
// ---------------------------------------------------------------------------

#define D64 64
#define RROWS 8192
#define TILE_ROWS 64   // rows per workgroup for GEMM-style kernels (4 waves x 16)

typedef __attribute__((ext_vector_type(16))) _Float16 v16h;
typedef __attribute__((ext_vector_type(8)))  _Float16 v8h;
typedef __attribute__((ext_vector_type(4)))  _Float16 v4h;
typedef __attribute__((ext_vector_type(8)))  float    v8f;
typedef __attribute__((ext_vector_type(4)))  float    v4f;

__device__ __forceinline__ int laneid() { return threadIdx.x & 31; }
__device__ __forceinline__ int waveid() { return threadIdx.x >> 5; }

__device__ __forceinline__ float fast_tanh(float x) {
#if __has_builtin(__builtin_amdgcn_tanhf)
  return __builtin_amdgcn_tanhf(x);
#elif __has_builtin(__builtin_amdgcn_tanh_f32)
  return __builtin_amdgcn_tanh_f32(x);
#else
  // 1 - 2/(e^{2x}+1): single native exp + rcp
  float e = __expf(2.f * x);
  return 1.f - 2.f * __frcp_rn(e + 1.f);
#endif
}

__device__ __forceinline__ float fast_sigmoid(float x) {
  return __frcp_rn(1.f + __expf(-x));
}

// A-fragment (16x32 f16, MxK) from row-major LDS tile.
__device__ __forceinline__ v16h lds_a_frag(const _Float16* base, int stride, int k0) {
  const int l = laneid();
  const _Float16* p = base + (l & 15) * stride + k0 + ((l >> 4) << 3);
  v8h lo = *(const v8h*)p;
  v8h hi = *(const v8h*)(p + 16);
  return __builtin_shufflevector(lo, hi, 0, 1, 2, 3, 4, 5, 6, 7,
                                 8, 9, 10, 11, 12, 13, 14, 15);
}

// Swizzled-B layout: element (k,n) of a KxN(=64) matrix lives at
// ((ktile*4 + ntile)*32 + lane)*16 + ((k&7) | (((k>>4)&1)<<3)),  lane = (n&15) | (((k>>3)&1)<<4)
__device__ __forceinline__ int b_swz_off(int k, int n) {
  return ((((k >> 5) << 2) + (n >> 4)) * 32 + ((n & 15) | (((k >> 3) & 1) << 4))) * 16
         + ((k & 7) | (((k >> 4) & 1) << 3));
}

// B-fragment (32x16 f16, KxN): 32B contiguous per lane -> two ds_load_b128.
__device__ __forceinline__ v16h ldsw_b_frag(const _Float16* base, int ktile, int ntile) {
  const _Float16* p = base + (((ktile << 2) + ntile) * 32 + laneid()) * 16;
  v8h lo = *(const v8h*)p;
  v8h hi = *(const v8h*)(p + 8);
  return __builtin_shufflevector(lo, hi, 0, 1, 2, 3, 4, 5, 6, 7,
                                 8, 9, 10, 11, 12, 13, 14, 15);
}

// C/D fragment element (m,n): vgpr r, lane l -> m = r + 8*(l>>4), n = l&15
__device__ __forceinline__ int frag_m(int r) { return r + ((laneid() >> 4) << 3); }
__device__ __forceinline__ int frag_n()      { return laneid() & 15; }

__device__ __forceinline__ v8f wmma_f16(v16h a, v16h b, v8f c) {
  return __builtin_amdgcn_wmma_f32_16x16x32_f16(false, a, false, b, (short)0, c,
                                                false, false);
}

// vectorized f32(global) -> f16(LDS) staging, count % 4 == 0
__device__ __forceinline__ void stage_f16(const float* __restrict__ src,
                                          _Float16* dst, int count, int tid, int nthr) {
  for (int i = tid * 4; i < count; i += nthr * 4) {
    v4f v = *(const v4f*)(src + i);
    v4h h;
#pragma unroll
    for (int q = 0; q < 4; ++q) h[q] = (_Float16)v[q];
    *(v4h*)(dst + i) = h;
  }
}

// stage a row-major KxN(=64) f32 weight into swizzled-B f16 LDS
__device__ __forceinline__ void stage_b_swz(const float* __restrict__ W,
                                            _Float16* dst, int count, int tid, int nthr) {
  for (int i = tid * 4; i < count; i += nthr * 4) {
    const int k = i >> 6, n = i & 63;
    v4f v = *(const v4f*)(W + i);
#pragma unroll
    for (int q = 0; q < 4; ++q) dst[b_swz_off(k, n + q)] = (_Float16)v[q];
  }
}

// ---------------------------------------------------------------------------
// Kernel 1: curv = tanh(x@Wc+bc)*0.1 ; field-norm xn ; det = xn@Wd+bd ;
//           coef = tanh([det,curv]@Wcoef + bcoef).  grid 128 x 128
// ---------------------------------------------------------------------------
__global__ __launch_bounds__(128) void k_prep(
    const float* __restrict__ x,  const float* __restrict__ Wc, const float* __restrict__ bc,
    const float* __restrict__ gam, const float* __restrict__ bet,
    const float* __restrict__ Wd, const float* __restrict__ bd,
    const float* __restrict__ Wcoef, const float* __restrict__ bcoef,
    float* __restrict__ xn_o, float* __restrict__ det_o, float* __restrict__ coef_o) {
  __shared__ alignas(16) _Float16 sWc[D64 * D64];
  __shared__ alignas(16) _Float16 sWd[D64 * D64];
  __shared__ alignas(16) _Float16 sXh[TILE_ROWS * D64];
  __shared__ alignas(16) _Float16 sXn[TILE_ROWS * D64];
  __shared__ alignas(16) _Float16 sDt[TILE_ROWS * D64];
  __shared__ alignas(16) float    sCv[TILE_ROWS * D64];
  __shared__ float sbc[D64], sgam[D64], sbet[D64], sbd[D64];
  __shared__ float sWco[128 * 4];
  __shared__ float sbco[4];

  const int tid  = threadIdx.x;
  const int row0 = blockIdx.x * TILE_ROWS;

  stage_b_swz(Wc, sWc, D64 * D64, tid, 128);
  stage_b_swz(Wd, sWd, D64 * D64, tid, 128);
  stage_f16(x + row0 * D64, sXh, TILE_ROWS * D64, tid, 128);
  for (int i = tid; i < D64; i += 128) { sbc[i] = bc[i]; sgam[i] = gam[i]; sbet[i] = bet[i]; sbd[i] = bd[i]; }
  for (int i = tid; i < 512; i += 128) sWco[i] = Wcoef[i];
  if (tid < 4) sbco[tid] = bcoef[tid];
  __syncthreads();

  const int w = waveid(), m0 = w * 16;

  // curv
  {
    v8f acc[4];
#pragma unroll
    for (int nt = 0; nt < 4; ++nt)
#pragma unroll
      for (int r = 0; r < 8; ++r) acc[nt][r] = 0.f;
#pragma unroll
    for (int kt = 0; kt < 2; ++kt) {
      v16h a = lds_a_frag(sXh + m0 * D64, D64, kt * 32);
#pragma unroll
      for (int nt = 0; nt < 4; ++nt)
        acc[nt] = wmma_f16(a, ldsw_b_frag(sWc, kt, nt), acc[nt]);
    }
#pragma unroll
    for (int nt = 0; nt < 4; ++nt)
#pragma unroll
      for (int r = 0; r < 8; ++r) {
        int m = m0 + frag_m(r), n = nt * 16 + frag_n();
        sCv[m * D64 + n] = fast_tanh(acc[nt][r] + sbc[n]) * 0.1f;
      }
  }
  __syncthreads();

  // field normalization (one thread per row)
  if (tid < TILE_ROWS) {
    const int r = tid;
    const float* xr = x + (row0 + r) * D64;
    float mu = 0.f, var = 0.f, cn = 0.f;
    for (int j = 0; j < D64; j += 4) {
      v4f xv = *(const v4f*)(xr + j);
      mu += xv[0] + xv[1] + xv[2] + xv[3];
    }
    mu *= (1.f / 64.f);
    for (int j = 0; j < D64; j += 4) {
      v4f xv = *(const v4f*)(xr + j);
#pragma unroll
      for (int q = 0; q < 4; ++q) { float d = xv[q] - mu; var += d * d; }
      v4f cv = *(const v4f*)(sCv + r * D64 + j);
#pragma unroll
      for (int q = 0; q < 4; ++q) cn += cv[q] * cv[q];
    }
    var *= (1.f / 64.f);
    float inv = rsqrtf(var + 1e-6f);
    cn = sqrtf(cn * (1.f / 64.f) + 1e-6f);
    float damp = __frcp_rn(1.f + cn);
    for (int j = 0; j < D64; j += 4) {
      v4f xv = *(const v4f*)(xr + j);
      v4f res;
      v4h h;
#pragma unroll
      for (int q = 0; q < 4; ++q) {
        res[q] = sgam[j + q] * (xv[q] - mu) * inv * damp + sbet[j + q];
        h[q] = (_Float16)res[q];
      }
      *(v4f*)(xn_o + (row0 + r) * D64 + j) = res;
      *(v4h*)(sXn + r * D64 + j) = h;
    }
  }
  __syncthreads();

  // det
  {
    v8f acc[4];
#pragma unroll
    for (int nt = 0; nt < 4; ++nt)
#pragma unroll
      for (int r = 0; r < 8; ++r) acc[nt][r] = 0.f;
#pragma unroll
    for (int kt = 0; kt < 2; ++kt) {
      v16h a = lds_a_frag(sXn + m0 * D64, D64, kt * 32);
#pragma unroll
      for (int nt = 0; nt < 4; ++nt)
        acc[nt] = wmma_f16(a, ldsw_b_frag(sWd, kt, nt), acc[nt]);
    }
#pragma unroll
    for (int nt = 0; nt < 4; ++nt)
#pragma unroll
      for (int r = 0; r < 8; ++r) {
        int m = m0 + frag_m(r), n = nt * 16 + frag_n();
        float v = acc[nt][r] + sbd[n];
        sDt[m * D64 + n] = (_Float16)v;
        det_o[(row0 + m) * D64 + n] = v;
      }
  }
  __syncthreads();

  // coef = tanh([det curv] @ Wcoef + b)
  if (tid < TILE_ROWS) {
    const int r = tid;
#pragma unroll
    for (int g = 0; g < 4; ++g) {
      float a = sbco[g];
      for (int j = 0; j < D64; ++j)
        a += (float)sDt[r * D64 + j] * sWco[j * 4 + g] + sCv[r * D64 + j] * sWco[(64 + j) * 4 + g];
      coef_o[(row0 + r) * 4 + g] = fast_tanh(a);
    }
  }
}

// ---------------------------------------------------------------------------
// Kernel 2: depthwise 3x3 SAME conv (mode 0); mode 1: BN + (-xn) -> ctx
// ---------------------------------------------------------------------------
__global__ __launch_bounds__(64) void k_dwconv(
    const float* __restrict__ in, const float* __restrict__ k9,
    const float* __restrict__ bng, const float* __restrict__ bnb,
    const float* __restrict__ sub, float* __restrict__ out, int mode) {
  const int p = blockIdx.x, d = threadIdx.x;
  const int wI = p & 31, h = (p >> 5) & 31, b = p >> 10;
  float acc = 0.f;
#pragma unroll
  for (int kh = 0; kh < 3; ++kh) {
    int hh = h + kh - 1;
    if (hh < 0 || hh > 31) continue;
#pragma unroll
    for (int kk = 0; kk < 3; ++kk) {
      int ww = wI + kk - 1;
      if (ww < 0 || ww > 31) continue;
      acc += in[((b * 32 + hh) * 32 + ww) * D64 + d] * k9[(kh * 3 + kk) * D64 + d];
    }
  }
  if (mode == 1) acc = bng[d] * acc + bnb[d] - sub[p * D64 + d];
  out[p * D64 + d] = acc;
}

// ---------------------------------------------------------------------------
// Kernel 3: ctx_t = ctx + A ctx ; skip = 3x (s -= 0.1 * A s)
// rank-4 decomposition: (v A^T) = 0.1 * sum_g coef_g (v Ga_g^T). grid 128x128
// ---------------------------------------------------------------------------
__global__ __launch_bounds__(128) void k_transport(
    const float* __restrict__ G, const float* __restrict__ coef,
    const float* __restrict__ ctx, const float* __restrict__ x,
    float* __restrict__ ctxt_o, float* __restrict__ skip_o) {
  __shared__ alignas(16) _Float16 sGaT[4 * D64 * D64];  // swizzled-B, [g] blocks
  __shared__ alignas(16) _Float16 sV[TILE_ROWS * D64];
  __shared__ float sCf[TILE_ROWS * 4];
  const int tid  = threadIdx.x;
  const int row0 = blockIdx.x * TILE_ROWS;

  for (int i = tid; i < 4 * 4096; i += 128) {
    int g = i >> 12, t = i & 4095, j = t >> 6, ii = t & 63;
    float v = G[g * 4096 + ii * 64 + j] - G[g * 4096 + j * 64 + ii];  // Ga[ii][j]
    sGaT[g * 4096 + b_swz_off(j, ii)] = (_Float16)v;                  // B[k=j][n=ii]
  }
  stage_f16(ctx + row0 * D64, sV, TILE_ROWS * D64, tid, 128);
  for (int i = tid; i < TILE_ROWS * 4; i += 128) sCf[i] = coef[row0 * 4 + i];
  __syncthreads();

  const int w = waveid(), m0 = w * 16;

  // ctx_t
  {
    v8f acc[4];
#pragma unroll
    for (int nt = 0; nt < 4; ++nt)
#pragma unroll
      for (int r = 0; r < 8; ++r)
        acc[nt][r] = ctx[(row0 + m0 + frag_m(r)) * D64 + nt * 16 + frag_n()];
#pragma unroll
    for (int g = 0; g < 4; ++g) {
      v8f u[4];
#pragma unroll
      for (int nt = 0; nt < 4; ++nt)
#pragma unroll
        for (int r = 0; r < 8; ++r) u[nt][r] = 0.f;
#pragma unroll
      for (int kt = 0; kt < 2; ++kt) {
        v16h a = lds_a_frag(sV + m0 * D64, D64, kt * 32);
#pragma unroll
        for (int nt = 0; nt < 4; ++nt)
          u[nt] = wmma_f16(a, ldsw_b_frag(sGaT + g * 4096, kt, nt), u[nt]);
      }
#pragma unroll
      for (int nt = 0; nt < 4; ++nt)
#pragma unroll
        for (int r = 0; r < 8; ++r)
          acc[nt][r] += 0.1f * sCf[(m0 + frag_m(r)) * 4 + g] * u[nt][r];
    }
#pragma unroll
    for (int nt = 0; nt < 4; ++nt)
#pragma unroll
      for (int r = 0; r < 8; ++r)
        ctxt_o[(row0 + m0 + frag_m(r)) * D64 + nt * 16 + frag_n()] = acc[nt][r];
  }
  __syncthreads();

  // skip transport (3 steps)
  {
    v8f s[4];
#pragma unroll
    for (int nt = 0; nt < 4; ++nt)
#pragma unroll
      for (int r = 0; r < 8; ++r)
        s[nt][r] = x[(row0 + m0 + frag_m(r)) * D64 + nt * 16 + frag_n()];

    for (int it = 0; it < 3; ++it) {
#pragma unroll
      for (int nt = 0; nt < 4; ++nt)
#pragma unroll
        for (int r = 0; r < 8; ++r)
          sV[(m0 + frag_m(r)) * D64 + nt * 16 + frag_n()] = (_Float16)s[nt][r];
      __syncthreads();
#pragma unroll
      for (int g = 0; g < 4; ++g) {
        v8f u[4];
#pragma unroll
        for (int nt = 0; nt < 4; ++nt)
#pragma unroll
          for (int r = 0; r < 8; ++r) u[nt][r] = 0.f;
#pragma unroll
        for (int kt = 0; kt < 2; ++kt) {
          v16h a = lds_a_frag(sV + m0 * D64, D64, kt * 32);
#pragma unroll
          for (int nt = 0; nt < 4; ++nt)
            u[nt] = wmma_f16(a, ldsw_b_frag(sGaT + g * 4096, kt, nt), u[nt]);
        }
#pragma unroll
        for (int nt = 0; nt < 4; ++nt)
#pragma unroll
          for (int r = 0; r < 8; ++r)
            s[nt][r] -= 0.01f * sCf[(m0 + frag_m(r)) * 4 + g] * u[nt][r];
      }
      __syncthreads();
    }
#pragma unroll
    for (int nt = 0; nt < 4; ++nt)
#pragma unroll
      for (int r = 0; r < 8; ++r)
        skip_o[(row0 + m0 + frag_m(r)) * D64 + nt * 16 + frag_n()] = s[nt][r];
  }
}

// ---------------------------------------------------------------------------
// Kernel 4: geo = rolling-product-terms(det) @ Wgeo + bgeo (K=384). grid 128x128
// ---------------------------------------------------------------------------
__device__ __forceinline__ float geo_term(const _Float16* __restrict__ dr, int c) {
  const int pair = c >> 6, j = c & 63, s = 1 << (pair >> 1);  // shifts 1,2,4
  const float dj = (float)dr[j];
  float v = dj * (float)dr[(j - s) & 63];
  if (pair & 1) v -= (float)dr[(j + s) & 63] * dj;
  return v;
}

__global__ __launch_bounds__(128) void k_geo(
    const float* __restrict__ det, const float* __restrict__ Wg,
    const float* __restrict__ bg, float* __restrict__ geo_o) {
  __shared__ alignas(16) _Float16 sWg[384 * D64];  // swizzled-B, 12 ktiles
  __shared__ alignas(16) _Float16 sDt[TILE_ROWS * D64];
  __shared__ float sbg[D64];
  const int tid  = threadIdx.x;
  const int row0 = blockIdx.x * TILE_ROWS;
  stage_b_swz(Wg, sWg, 384 * D64, tid, 128);
  stage_f16(det + row0 * D64, sDt, TILE_ROWS * D64, tid, 128);
  for (int i = tid; i < D64; i += 128) sbg[i] = bg[i];
  __syncthreads();

  const int w = waveid(), m0 = w * 16, l = laneid();
  const _Float16* dr = sDt + (m0 + (l & 15)) * D64;
  const int kbase = (l >> 4) << 3;

  v8f acc[4];
#pragma unroll
  for (int nt = 0; nt < 4; ++nt)
#pragma unroll
    for (int r = 0; r < 8; ++r) acc[nt][r] = 0.f;

  for (int kt = 0; kt < 12; ++kt) {
    const int k0 = kt * 32 + kbase;
    v16h a;
#pragma unroll
    for (int i = 0; i < 8; ++i) {
      a[i]     = (_Float16)geo_term(dr, k0 + i);
      a[8 + i] = (_Float16)geo_term(dr, k0 + 16 + i);
    }
#pragma unroll
    for (int nt = 0; nt < 4; ++nt)
      acc[nt] = wmma_f16(a, ldsw_b_frag(sWg, kt, nt), acc[nt]);
  }
#pragma unroll
  for (int nt = 0; nt < 4; ++nt)
#pragma unroll
    for (int r = 0; r < 8; ++r) {
      int m = m0 + frag_m(r), n = nt * 16 + frag_n();
      geo_o[(row0 + m) * D64 + n] = acc[nt][r] + sbg[n];
    }
}

// ---------------------------------------------------------------------------
// Kernel 5: holonomy. M = prod_{t=0..7}(I + A_{s+t}) via WMMA chain,
// traces at product lengths {2,4,8}. 16 positions per block. grid 512x128
// N is synthesized in packed f16 (v_pk_fma_f16) with an LDS identity table.
// ---------------------------------------------------------------------------
__global__ __launch_bounds__(128) void k_holo(
    const float* __restrict__ G, const float* __restrict__ coef,
    float* __restrict__ hol_raw) {
  __shared__ alignas(16) _Float16 sGa[4 * 4096];  // row-major Ga_g[i][j]
  __shared__ alignas(16) _Float16 sId[4096];      // identity, row-major
  __shared__ alignas(16) _Float16 sM[4096];       // row-major (A operand)
  __shared__ alignas(16) _Float16 sN[4096];       // swizzled-B (B operand)
  __shared__ float sRed[128];
  const int tid = threadIdx.x;
  for (int i = tid * 4; i < 4 * 4096; i += 512) {
    const int g = i >> 12, t = i & 4095, ii = t >> 6, j = t & 63;
    v4f a = *(const v4f*)(G + g * 4096 + ii * 64 + j);
    v4h h;
#pragma unroll
    for (int q = 0; q < 4; ++q) h[q] = (_Float16)(a[q] - G[g * 4096 + (j + q) * 64 + ii]);
    *(v4h*)(sGa + i) = h;
  }
  for (int i = tid * 4; i < 4096; i += 512) {
    const int ii = i >> 6, j = i & 63;
    v4h h;
#pragma unroll
    for (int q = 0; q < 4; ++q) h[q] = (_Float16)((ii == j + q) ? 1.f : 0.f);
    *(v4h*)(sId + i) = h;
  }
  __syncthreads();

  const int w = waveid(), m0 = w * 16, lane = laneid();

  for (int p = 0; p < 16; ++p) {
    const int row  = blockIdx.x * 16 + p;
    const int b    = row >> 10;
    const int sIdx = row & 1023;

    // M = I (fragment registers)
    v8f acc[4];
#pragma unroll
    for (int nt = 0; nt < 4; ++nt)
#pragma unroll
      for (int r = 0; r < 8; ++r)
        acc[nt][r] = (m0 + frag_m(r) == nt * 16 + frag_n()) ? 1.f : 0.f;

    for (int t = 0; t < 8; ++t) {
      const int s2 = (sIdx + t) & 1023;
      const float* cf = coef + (b * 1024 + s2) * 4;
      if (t < 7) __builtin_prefetch(coef + (b * 1024 + ((sIdx + t + 1) & 1023)) * 4, 0, 1);
      const _Float16 c0 = (_Float16)(cf[0] * 0.1f), c1 = (_Float16)(cf[1] * 0.1f);
      const _Float16 c2 = (_Float16)(cf[2] * 0.1f), c3 = (_Float16)(cf[3] * 0.1f);
      // build N = I + A_{s+t} into swizzled-B LDS (packed f16 FMAs)
      for (int base = tid * 8; base < 4096; base += 1024) {
        const int j = base >> 6, k = base & 63;
        v8h n8 = *(const v8h*)(sId + base);
        n8 += *(const v8h*)(sGa + base) * c0;
        n8 += *(const v8h*)(sGa + 4096 + base) * c1;
        n8 += *(const v8h*)(sGa + 8192 + base) * c2;
        n8 += *(const v8h*)(sGa + 12288 + base) * c3;
#pragma unroll
        for (int q = 0; q < 8; ++q) sN[b_swz_off(j, k + q)] = n8[q];
      }
      // stage current M rows owned by this wave (row-major, A operand)
#pragma unroll
      for (int nt = 0; nt < 4; ++nt)
#pragma unroll
        for (int r = 0; r < 8; ++r)
          sM[(m0 + frag_m(r)) * D64 + nt * 16 + frag_n()] = (_Float16)acc[nt][r];
      __syncthreads();

      v8f nx[4];
#pragma unroll
      for (int nt = 0; nt < 4; ++nt)
#pragma unroll
        for (int r = 0; r < 8; ++r) nx[nt][r] = 0.f;
#pragma unroll
      for (int kt = 0; kt < 2; ++kt) {
        v16h a = lds_a_frag(sM + m0 * D64, D64, kt * 32);
#pragma unroll
        for (int nt = 0; nt < 4; ++nt)
          nx[nt] = wmma_f16(a, ldsw_b_frag(sN, kt, nt), nx[nt]);
      }
#pragma unroll
      for (int nt = 0; nt < 4; ++nt) acc[nt] = nx[nt];

      if (t == 1 || t == 3 || t == 7) {  // product lengths 2,4,8
        const int li = (t == 1) ? 0 : ((t == 3) ? 1 : 2);
        float part = 0.f;
#pragma unroll
        for (int nt = 0; nt < 4; ++nt)
          if (nt == w) {
#pragma unroll
            for (int r = 0; r < 8; ++r) {
              if (lane == r) part += acc[nt][r];       // lanes 0..7  (cols 0-7)
              if (lane - 24 == r) part += acc[nt][r];  // lanes 24..31 (cols 8-15)
            }
          }
        sRed[tid] = part;
        __syncthreads();
        if (tid == 0) {
          float ssum = 0.f;
          for (int i = 0; i < 128; ++i) ssum += sRed[i];
          atomicAdd(&hol_raw[b * 3 + li], ssum * (1.f / (64.f * 1024.f)));
        }
      }
      __syncthreads();
    }
  }
}

// ---------------------------------------------------------------------------
// Kernel 6: hol_ctx[b,d] = bhol[d] + sum_t hol[b,t] * Whol[t,d]
// ---------------------------------------------------------------------------
__global__ __launch_bounds__(256) void k_holctx(
    const float* __restrict__ hol_raw, const float* __restrict__ Wh,
    const float* __restrict__ bh, float* __restrict__ hctx) {
  const int i = blockIdx.x * 256 + threadIdx.x;
  if (i >= 8 * 64) return;
  const int b = i >> 6, d = i & 63;
  float v = bh[d];
#pragma unroll
  for (int t = 0; t < 3; ++t) v += hol_raw[b * 3 + t] * Wh[t * 64 + d];
  hctx[i] = v;
}

// ---------------------------------------------------------------------------
// Kernel 7: stream = geo + ctx_t + hol_ctx;
// out = skip + layer_scale * sigmoid(stream@Wgate + bgate) * stream. grid 128x128
// ---------------------------------------------------------------------------
__global__ __launch_bounds__(128) void k_final(
    const float* __restrict__ geo, const float* __restrict__ ctxt,
    const float* __restrict__ hctx, const float* __restrict__ skip,
    const float* __restrict__ Wg, const float* __restrict__ bg,
    const float* __restrict__ ls, float* __restrict__ out) {
  __shared__ alignas(16) _Float16 sW[D64 * D64];   // swizzled-B
  __shared__ alignas(16) _Float16 sS[TILE_ROWS * D64];
  __shared__ alignas(16) float    sS32[TILE_ROWS * D64];
  __shared__ float sbg[D64], sls[D64];
  const int tid  = threadIdx.x;
  const int row0 = blockIdx.x * TILE_ROWS;
  stage_b_swz(Wg, sW, D64 * D64, tid, 128);
  for (int i = tid; i < D64; i += 128) { sbg[i] = bg[i]; sls[i] = ls[i]; }
  for (int i = tid * 4; i < TILE_ROWS * D64; i += 512) {
    const int gr = row0 + (i >> 6), d = i & 63;
    v4f gv = *(const v4f*)(geo + row0 * D64 + i);
    v4f cv = *(const v4f*)(ctxt + row0 * D64 + i);
    v4f hv = *(const v4f*)(hctx + (gr >> 10) * D64 + d);
    v4f s = gv + cv + hv;
    *(v4f*)(sS32 + i) = s;
    v4h h;
#pragma unroll
    for (int q = 0; q < 4; ++q) h[q] = (_Float16)s[q];
    *(v4h*)(sS + i) = h;
  }
  __syncthreads();

  const int w = waveid(), m0 = w * 16;
  v8f acc[4];
#pragma unroll
  for (int nt = 0; nt < 4; ++nt)
#pragma unroll
    for (int r = 0; r < 8; ++r) acc[nt][r] = 0.f;
#pragma unroll
  for (int kt = 0; kt < 2; ++kt) {
    v16h a = lds_a_frag(sS + m0 * D64, D64, kt * 32);
#pragma unroll
    for (int nt = 0; nt < 4; ++nt)
      acc[nt] = wmma_f16(a, ldsw_b_frag(sW, kt, nt), acc[nt]);
  }
#pragma unroll
  for (int nt = 0; nt < 4; ++nt)
#pragma unroll
    for (int r = 0; r < 8; ++r) {
      const int m = m0 + frag_m(r), n = nt * 16 + frag_n();
      const float gate = fast_sigmoid(acc[nt][r] + sbg[n]);
      const float sv = sS32[m * D64 + n];
      const int gi = (row0 + m) * D64 + n;
      out[gi] = skip[gi] + sls[n] * gate * sv;
    }
}

__global__ void k_zero(float* __restrict__ p, int n) {
  const int i = blockIdx.x * 64 + threadIdx.x;
  if (i < n) p[i] = 0.f;
}

// ---------------------------------------------------------------------------
extern "C" void kernel_launch(void* const* d_in, const int* in_sizes, int n_in,
                              void* d_out, int out_size, void* d_ws, size_t ws_size,
                              hipStream_t stream) {
  (void)in_sizes; (void)n_in; (void)out_size; (void)ws_size;
  const float* x     = (const float*)d_in[0];
  const float* Wc    = (const float*)d_in[1];
  const float* bc    = (const float*)d_in[2];
  const float* gam   = (const float*)d_in[3];
  const float* bet   = (const float*)d_in[4];
  const float* Wd    = (const float*)d_in[5];
  const float* bd    = (const float*)d_in[6];
  const float* dw1   = (const float*)d_in[7];
  const float* dw2   = (const float*)d_in[8];
  const float* bng   = (const float*)d_in[9];
  const float* bnb   = (const float*)d_in[10];
  const float* Wcoef = (const float*)d_in[11];
  const float* bcoef = (const float*)d_in[12];
  const float* G     = (const float*)d_in[13];
  const float* Wgeo  = (const float*)d_in[14];
  const float* bgeo  = (const float*)d_in[15];
  const float* Whol  = (const float*)d_in[16];
  const float* bhol  = (const float*)d_in[17];
  const float* Wgate = (const float*)d_in[18];
  const float* bgate = (const float*)d_in[19];
  const float* ls    = (const float*)d_in[20];

  float* w = (float*)d_ws;
  const size_t T = (size_t)RROWS * D64;   // 524288 floats per tensor
  float* xn   = w;
  float* det  = w + T;
  float* ctx  = w + 2 * T;
  float* ctxt = w + 3 * T;
  float* skp  = w + 4 * T;
  float* tmp  = w + 5 * T;                // dwconv intermediate, reused as geo
  float* geo  = tmp;                      // alias: tmp dead before geo written
  float* coef = w + 6 * T;                // 32768 floats
  float* hraw = coef + 32768;             // 24 floats
  float* hctx = hraw + 32;                // 512 floats

  k_zero<<<1, 64, 0, stream>>>(hraw, 24);
  k_prep<<<128, 128, 0, stream>>>(x, Wc, bc, gam, bet, Wd, bd, Wcoef, bcoef,
                                  xn, det, coef);
  k_dwconv<<<8192, 64, 0, stream>>>(xn, dw1, bng, bnb, xn, tmp, 0);
  k_dwconv<<<8192, 64, 0, stream>>>(tmp, dw2, bng, bnb, xn, ctx, 1);
  k_transport<<<128, 128, 0, stream>>>(G, coef, ctx, x, ctxt, skp);
  k_holo<<<512, 128, 0, stream>>>(G, coef, hraw);
  k_geo<<<128, 128, 0, stream>>>(det, Wgeo, bgeo, geo);
  k_holctx<<<2, 256, 0, stream>>>(hraw, Whol, bhol, hctx);
  k_final<<<128, 128, 0, stream>>>(geo, ctxt, hctx, skp, Wgate, bgate, ls,
                                   (float*)d_out);
}
